// MaskedGenerativeEncoderViT_15659450761453
// MI455X (gfx1250) — matrix operand order repaired and checked by reading, not compiled, not run on hardware
//
#include <hip/hip_runtime.h>
#include <hip/hip_bf16.h>

// ---------------------------------------------------------------------------
// Problem constants
// ---------------------------------------------------------------------------
#define BATCH   8
#define GRID_N  512
#define DMODEL  1024
#define NHEAD   16
#define HEADD   64
#define NLAYER  8
#define FFDIM   4096
#define MROWS   (BATCH * GRID_N)      // 4096
#define TRUE_LBL  512
#define FALSE_LBL 513
#define MASK_LBL  514

typedef __bf16 bf16_t;
typedef __attribute__((ext_vector_type(16))) __bf16 v16bf;
typedef __attribute__((ext_vector_type(8)))  float  v8f;
typedef __attribute__((ext_vector_type(4)))  unsigned int v4u;
typedef __attribute__((ext_vector_type(4)))  int v4i;
typedef __attribute__((ext_vector_type(8)))  int v8i;

// Tensor Data Mover path: probe-verified 5-arg builtin on ROCm 7.2 (clang-22).
#if defined(__has_builtin)
#  if __has_builtin(__builtin_amdgcn_tensor_load_to_lds) && (__clang_major__ < 23)
#    define USE_TDM 1
#  endif
#endif
#ifndef USE_TDM
#  define USE_TDM 0
#endif

// VALU lane-permute reductions (v_permlane16_b32 / v_permlanex16_b32)
#if defined(__has_builtin)
#  if __has_builtin(__builtin_amdgcn_permlane16) && __has_builtin(__builtin_amdgcn_permlanex16)
#    define USE_PERMLANE 1
#  endif
#endif
#ifndef USE_PERMLANE
#  define USE_PERMLANE 0
#endif

#define ATH_STR2(x) #x
#define ATH_STR(x) ATH_STR2(x)
#pragma message("ATHENA probe: clang=" ATH_STR(__clang_major__) " USE_TDM=" ATH_STR(USE_TDM) " USE_PERMLANE=" ATH_STR(USE_PERMLANE))

union Frag {            // one WMMA 16-half operand: 8 VGPRs = 2 x 16B chunks
    v16bf v;
    uint4 q[2];
};

__device__ __forceinline__ unsigned short f2bf(float f) {
    unsigned u = __float_as_uint(f);
    unsigned r = u + 0x7FFFu + ((u >> 16) & 1u);      // round-to-nearest-even
    if ((u & 0x7F800000u) == 0x7F800000u) r = u;      // keep inf/nan payload
    return (unsigned short)(r >> 16);
}

// ---------------------------------------------------------------------------
// Lane-permute helpers: XOR-butterfly within rows of 16 lanes (VALU, no LDS),
// and the cross-half (lane^16) exchange for full-wave reductions.
// ---------------------------------------------------------------------------
template <int M>
__device__ __forceinline__ float xor16f(float x) {
#if USE_PERMLANE
    unsigned s0, s1;
    if      (M == 1) { s0 = 0x67452301u; s1 = 0xEFCDAB89u; }
    else if (M == 2) { s0 = 0x54761032u; s1 = 0xDCFE98BAu; }
    else if (M == 4) { s0 = 0x32107654u; s1 = 0xBA98FEDCu; }
    else             { s0 = 0xFEDCBA98u; s1 = 0x76543210u; }   // M == 8
    const unsigned u = __float_as_uint(x);
    return __uint_as_float(
        __builtin_amdgcn_permlane16(u, u, s0, s1, false, false));
#else
    return __shfl_xor(x, M, 16);
#endif
}

__device__ __forceinline__ float xhalf32f(float x) {           // lane ^ 16
#if USE_PERMLANE
    const unsigned u = __float_as_uint(x);
    return __uint_as_float(
        __builtin_amdgcn_permlanex16(u, u, 0x76543210u, 0xFEDCBA98u, false, false));
#else
    return __shfl_xor(x, 16, 32);
#endif
}

__device__ __forceinline__ float rmax16(float x) {
    x = fmaxf(x, xor16f<1>(x)); x = fmaxf(x, xor16f<2>(x));
    x = fmaxf(x, xor16f<4>(x)); x = fmaxf(x, xor16f<8>(x));
    return x;
}
__device__ __forceinline__ float rsum16(float x) {
    x += xor16f<1>(x); x += xor16f<2>(x);
    x += xor16f<4>(x); x += xor16f<8>(x);
    return x;
}
__device__ __forceinline__ float rsum32(float x) {
    x = rsum16(x);
    x += xhalf32f(x);
    return x;
}

#if USE_TDM
// ---------------------------------------------------------------------------
// Issue one TDM 2D tile load: tile_d1 rows x tile_d0 bf16 elements, row pitch
// `pitch` elements in global memory, packed row-major into LDS at lds_off.
// D# packing per CDNA5 ISA 8.3/8.4 (group0 128b, group1 256b, groups2/3 = 0).
// ---------------------------------------------------------------------------
__device__ __forceinline__ void tdm_load_tile_2d(
    unsigned lds_off, const unsigned short* gptr,
    unsigned tile_d0, unsigned tile_d1, unsigned pitch)
{
    const unsigned long long ga = (unsigned long long)(uintptr_t)gptr;
    v4u g0;
    g0[0] = 1u;                                          // count=1, user D#
    g0[1] = lds_off;                                     // lds_addr (bytes)
    g0[2] = (unsigned)(ga & 0xFFFFFFFFu);                // global_addr[31:0]
    g0[3] = (unsigned)((ga >> 32) & 0x01FFFFFFu)         // global_addr[56:32]
          | (2u << 30);                                  // type = 2 ("image")
    v8i g1;
    g1[0] = (int)(1u << 16);                             // data_size=1 (2B)
    g1[1] = (int)((tile_d0 & 0xFFFFu) << 16);            // tensor_dim0[15:0]
    g1[2] = (int)(((tile_d0 >> 16) & 0xFFFFu)            // tensor_dim0[31:16]
          | ((tile_d1 & 0xFFFFu) << 16));                // tensor_dim1[15:0]
    g1[3] = (int)(((tile_d1 >> 16) & 0xFFFFu)            // tensor_dim1[31:16]
          | ((tile_d0 & 0xFFFFu) << 16));                // tile_dim0
    g1[4] = (int)(tile_d1 & 0xFFFFu);                    // tile_dim1 (tile_dim2=0)
    g1[5] = (int)pitch;                                  // tensor_dim0_stride[31:0]
    g1[6] = 0;                                           // stride hi / dim1_stride lo
    g1[7] = 0;
    const v4i zz = {0, 0, 0, 0};                         // 2D: groups 2/3 unused
    __builtin_amdgcn_tensor_load_to_lds(g0, g1, zz, zz, 0);
}
#endif

// ---------------------------------------------------------------------------
// GEMM:  out[M,N] = A_bf16[M,K] @ W_bf16[N,K]^T (+bias)(+resid)(gelu->bf16)
// BM=BN=128, BK=32, 256 threads = 8 waves (4 along M x 2 along N),
// each wave computes 32x64 = 2x4 v_wmma_f32_16x16x32_bf16 accumulators.
// LDS double-buffered; tiles DMA'd by the Tensor Data Mover (wave 0 issues
// next-step descriptors, s_wait_tensorcnt, workgroup barrier), with a
// synchronous b128 staging fallback when the TDM builtin is unavailable.
// EPI: 0 = bias -> fp32 ; 1 = bias+resid -> fp32 ; 2 = bias+GELU -> bf16
// ---------------------------------------------------------------------------
#define BM 128
#define BN 128
#define BK 32

template <int EPI>
__global__ __launch_bounds__(256) void gemm_bf16_wmma(
    const unsigned short* __restrict__ A,
    const unsigned short* __restrict__ W,
    const float*          __restrict__ bias,
    const float*          __restrict__ resid,
    float*                __restrict__ outF,
    unsigned short*       __restrict__ outBf,
    int M, int N, int K)
{
    __shared__ __align__(16) unsigned short As[2][BM * BK];
    __shared__ __align__(16) unsigned short Bs[2][BN * BK];

    const int tid  = threadIdx.x;
    const int lane = tid & 31;
    const int wid  = tid >> 5;
    const int wm   = wid & 3;          // 4 wave-rows   (32 rows each)
    const int wn   = wid >> 2;         // 2 wave-cols   (64 cols each)
    const int m0   = blockIdx.y * BM;
    const int n0   = blockIdx.x * BN;
    const int nl   = lane & 15;
    const int g    = lane >> 4;

    v8f acc[2][4];
#pragma unroll
    for (int i = 0; i < 2; ++i)
#pragma unroll
        for (int j = 0; j < 4; ++j)
#pragma unroll
            for (int e = 0; e < 8; ++e) acc[i][j][e] = 0.0f;

    const int nsteps = K / BK;

#if USE_TDM
    const unsigned asOff = (unsigned)(uintptr_t)&As[0][0];
    const unsigned bsOff = (unsigned)(uintptr_t)&Bs[0][0];
    if (wid == 0) {   // prologue: DMA tiles for step 0 into buffer 0
        tdm_load_tile_2d(asOff, A + (size_t)m0 * K, BK, BM, K);
        tdm_load_tile_2d(bsOff, W + (size_t)n0 * K, BK, BN, K);
    }
#else
    const int ldr = tid >> 1;          // 0..127 : tile row to stage
    const int ldc = (tid & 1) * 16;    // 0 / 16 : 16-element chunk
    {
        const unsigned short* gpA = A + (size_t)(m0 + ldr) * K + ldc;
        const unsigned short* gpB = W + (size_t)(n0 + ldr) * K + ldc;
        *(uint4*)&As[0][ldr * BK + ldc]     = *(const uint4*)gpA;
        *(uint4*)&As[0][ldr * BK + ldc + 8] = *(const uint4*)(gpA + 8);
        *(uint4*)&Bs[0][ldr * BK + ldc]     = *(const uint4*)gpB;
        *(uint4*)&Bs[0][ldr * BK + ldc + 8] = *(const uint4*)(gpB + 8);
    }
#endif

    for (int s = 0; s < nsteps; ++s) {
        const int buf  = s & 1;
        const bool more = (s + 1) < nsteps;
#if USE_TDM
        if (wid == 0) {
            if (more) {   // DMA next tiles into the other buffer
                const int k1 = (s + 1) * BK;
                tdm_load_tile_2d(asOff + (buf ^ 1) * (BM * BK * 2),
                                 A + (size_t)m0 * K + k1, BK, BM, K);
                tdm_load_tile_2d(bsOff + (buf ^ 1) * (BN * BK * 2),
                                 W + (size_t)n0 * K + k1, BK, BN, K);
                __builtin_amdgcn_s_wait_tensorcnt((short)2);  // current tiles done
            } else {
                __builtin_amdgcn_s_wait_tensorcnt((short)0);
            }
        }
        __syncthreads();          // current tiles visible to all waves
#else
        __syncthreads();
        if (more) {               // prefetch next tiles after barrier
            const int k1 = (s + 1) * BK;
            const unsigned short* gpA = A + (size_t)(m0 + ldr) * K + k1 + ldc;
            const unsigned short* gpB = W + (size_t)(n0 + ldr) * K + k1 + ldc;
            *(uint4*)&As[buf ^ 1][ldr * BK + ldc]     = *(const uint4*)gpA;
            *(uint4*)&As[buf ^ 1][ldr * BK + ldc + 8] = *(const uint4*)(gpA + 8);
            *(uint4*)&Bs[buf ^ 1][ldr * BK + ldc]     = *(const uint4*)gpB;
            *(uint4*)&Bs[buf ^ 1][ldr * BK + ldc + 8] = *(const uint4*)(gpB + 8);
        }
#endif
        // ---- fragments from LDS + 8 WMMAs
        Frag af[2];
#pragma unroll
        for (int i = 0; i < 2; ++i) {
            const int row = wm * 32 + i * 16 + nl;      // A: interleaved K layout
            af[i].q[0] = *(const uint4*)&As[buf][row * BK + g * 8];
            af[i].q[1] = *(const uint4*)&As[buf][row * BK + 16 + g * 8];
        }
        Frag bfg[4];
#pragma unroll
        for (int j = 0; j < 4; ++j) {
            const int col = wn * 64 + j * 16 + nl;      // B: interleaved K layout
            bfg[j].q[0] = *(const uint4*)&Bs[buf][col * BK + g * 8];
            bfg[j].q[1] = *(const uint4*)&Bs[buf][col * BK + 16 + g * 8];
        }
#pragma unroll
        for (int i = 0; i < 2; ++i)
#pragma unroll
            for (int j = 0; j < 4; ++j)
                acc[i][j] = __builtin_amdgcn_wmma_f32_16x16x32_bf16(
                    false, af[i].v, false, bfg[j].v, (short)0, acc[i][j],
                    false, false);
        __syncthreads();          // everyone done with buf before it is refilled
    }

    // ---- epilogue
#pragma unroll
    for (int i = 0; i < 2; ++i)
#pragma unroll
        for (int j = 0; j < 4; ++j) {
            const int nn = n0 + wn * 64 + j * 16 + nl;
            const float bv = bias[nn];
#pragma unroll
            for (int e = 0; e < 8; ++e) {
                const int mm = m0 + wm * 32 + i * 16 + e + 8 * g;
                float val = acc[i][j][e] + bv;
                if (EPI == 1) val += resid[(size_t)mm * N + nn];
                if (EPI == 2) {
                    val = 0.5f * val * (1.0f + erff(val * 0.70710678118f));
                    outBf[(size_t)mm * N + nn] = f2bf(val);
                } else {
                    outF[(size_t)mm * N + nn] = val;
                }
            }
        }
}

// ---------------------------------------------------------------------------
// fp32 -> bf16 bulk conversion (weights, once per launch)
// ---------------------------------------------------------------------------
__global__ __launch_bounds__(256) void conv_f32_bf16(
    const float* __restrict__ src, unsigned short* __restrict__ dst, int n4)
{
    const int i = blockIdx.x * 256 + threadIdx.x;
    if (i < n4) {
        const float4 f = *(const float4*)(src + (size_t)i * 4);
        const unsigned lo = (unsigned)f2bf(f.x) | ((unsigned)f2bf(f.y) << 16);
        const unsigned hi = (unsigned)f2bf(f.z) | ((unsigned)f2bf(f.w) << 16);
        uint2 v; v.x = lo; v.y = hi;
        *(uint2*)(dst + (size_t)i * 4) = v;
    }
}

// ---------------------------------------------------------------------------
// QKV reorg: QKVf[4096,3072] -> Q,K [bh,n,64] bf16 and V^T [bh,64,n] bf16
// ---------------------------------------------------------------------------
__global__ __launch_bounds__(256) void qkv_reorg(
    const float* __restrict__ QKV,
    unsigned short* __restrict__ Qb,
    unsigned short* __restrict__ Kb,
    unsigned short* __restrict__ Vt)
{
    const int idx = blockIdx.x * 256 + threadIdx.x;   // 4,194,304 total
    const int d = idx & 63;
    const int n = (idx >> 6) & 511;
    const int h = (idx >> 15) & 15;
    const int b = idx >> 19;
    const size_t row = (size_t)b * GRID_N + n;
    const int c = h * HEADD + d;
    const float q = QKV[row * (3 * DMODEL) + c];
    const float k = QKV[row * (3 * DMODEL) + DMODEL + c];
    const float v = QKV[row * (3 * DMODEL) + 2 * DMODEL + c];
    const int bh = b * NHEAD + h;
    Qb[((size_t)bh * GRID_N + n) * HEADD + d] = f2bf(q);
    Kb[((size_t)bh * GRID_N + n) * HEADD + d] = f2bf(k);
    Vt[((size_t)bh * HEADD + d) * GRID_N + n] = f2bf(v);
}

// ---------------------------------------------------------------------------
// Flash attention: 1 wave per (b,h, 16-query block). WMMA for QK^T and P*V,
// online softmax with permlane16 row reductions (VALU, matches C layout).
// ---------------------------------------------------------------------------
__global__ __launch_bounds__(256) void attn_flash(
    const unsigned short* __restrict__ Qb,
    const unsigned short* __restrict__ Kb,
    const unsigned short* __restrict__ Vt,
    unsigned short* __restrict__ Y)            // [B*N, D] bf16
{
    __shared__ __align__(16) unsigned short P[8][16 * 32];

    const int tid  = threadIdx.x;
    const int lane = tid & 31;
    const int wi   = tid >> 5;
    const int w    = blockIdx.x * 8 + wi;      // 0..4095
    const int bh   = w >> 5;                   // 0..127
    const int q0   = (w & 31) * 16;
    const int b    = bh >> 4;
    const int h    = bh & 15;
    const int nl   = lane & 15;
    const int g    = lane >> 4;
    const float scale = 0.125f;                // 64^-0.5

    Frag qf[2];
    {
        const unsigned short* qp =
            Qb + ((size_t)bh * GRID_N + q0 + nl) * HEADD;
#pragma unroll
        for (int kk = 0; kk < 2; ++kk) {
            qf[kk].q[0] = *(const uint4*)(qp + kk * 32 + g * 8);
            qf[kk].q[1] = *(const uint4*)(qp + kk * 32 + 16 + g * 8);
        }
    }

    v8f  o[4];
    float mrow[8], lrow[8];
#pragma unroll
    for (int t = 0; t < 4; ++t)
#pragma unroll
        for (int e = 0; e < 8; ++e) o[t][e] = 0.0f;
#pragma unroll
    for (int e = 0; e < 8; ++e) { mrow[e] = -1e30f; lrow[e] = 0.0f; }

    for (int kt = 0; kt < GRID_N / 32; ++kt) {
        const int key0 = kt * 32;

        v8f s[2];
#pragma unroll
        for (int t = 0; t < 2; ++t) {
#pragma unroll
            for (int e = 0; e < 8; ++e) s[t][e] = 0.0f;
            const unsigned short* kp =
                Kb + ((size_t)bh * GRID_N + key0 + t * 16 + nl) * HEADD;
#pragma unroll
            for (int kk = 0; kk < 2; ++kk) {
                Frag kf;                       // B layout: interleaved d per half
                kf.q[0] = *(const uint4*)(kp + kk * 32 + g * 8);
                kf.q[1] = *(const uint4*)(kp + kk * 32 + 16 + g * 8);
                s[t] = __builtin_amdgcn_wmma_f32_16x16x32_bf16(
                    false, qf[kk].v, false, kf.v, (short)0, s[t], false, false);
            }
        }

#pragma unroll
        for (int e = 0; e < 8; ++e) {
            const float a0 = s[0][e] * scale;
            const float a1 = s[1][e] * scale;
            const float tmax  = rmax16(fmaxf(a0, a1));
            const float nm    = fmaxf(mrow[e], tmax);
            const float alpha = __expf(mrow[e] - nm);
            const float p0 = __expf(a0 - nm);
            const float p1 = __expf(a1 - nm);
            const float ps = rsum16(p0 + p1);
            lrow[e] = lrow[e] * alpha + ps;
            mrow[e] = nm;
#pragma unroll
            for (int t = 0; t < 4; ++t) o[t][e] *= alpha;
            const int r = e + 8 * g;           // C-layout row held by this lane
            P[wi][r * 32 + nl]      = f2bf(p0);
            P[wi][r * 32 + 16 + nl] = f2bf(p1);
        }

        Frag pf;
        pf.q[0] = *(const uint4*)&P[wi][nl * 32 + g * 8];
        pf.q[1] = *(const uint4*)&P[wi][nl * 32 + 16 + g * 8];
#pragma unroll
        for (int t = 0; t < 4; ++t) {
            const unsigned short* vp =
                Vt + ((size_t)bh * HEADD + t * 16 + nl) * GRID_N + key0 + g * 8;
            Frag vf;
            vf.q[0] = *(const uint4*)vp;
            vf.q[1] = *(const uint4*)(vp + 16);
            o[t] = __builtin_amdgcn_wmma_f32_16x16x32_bf16(
                false, pf.v, false, vf.v, (short)0, o[t], false, false);
        }
    }

#pragma unroll
    for (int e = 0; e < 8; ++e) {
        const float inv = 1.0f / lrow[e];
        const int  q    = q0 + e + 8 * g;
#pragma unroll
        for (int t = 0; t < 4; ++t) {
            const int d = t * 16 + nl;
            Y[((size_t)b * GRID_N + q) * DMODEL + h * HEADD + d] =
                f2bf(o[t][e] * inv);
        }
    }
}

// ---------------------------------------------------------------------------
// LayerNorm (fp32 in) -> bf16 out. One wave per 1024-wide row.
// ---------------------------------------------------------------------------
__global__ __launch_bounds__(256) void ln_to_bf16(
    const float* __restrict__ X,
    const float* __restrict__ w,
    const float* __restrict__ bsh,
    float eps,
    unsigned short* __restrict__ out)
{
    const int lane = threadIdx.x & 31;
    const int row  = blockIdx.x * 8 + (threadIdx.x >> 5);
    const float* xr = X + (size_t)row * DMODEL;
    float v[32], s = 0.0f, ss = 0.0f;
#pragma unroll
    for (int i = 0; i < 32; ++i) {
        const float x = xr[i * 32 + lane];
        v[i] = x; s += x; ss += x * x;
    }
    s  = rsum32(s);
    ss = rsum32(ss);
    const float mu  = s * (1.0f / DMODEL);
    const float var = ss * (1.0f / DMODEL) - mu * mu;
    const float rs  = rsqrtf(var + eps);
#pragma unroll
    for (int i = 0; i < 32; ++i) {
        const int d = i * 32 + lane;
        out[(size_t)row * DMODEL + d] = f2bf((v[i] - mu) * rs * w[d] + bsh[d]);
    }
}

// ---------------------------------------------------------------------------
// Embedding gather + LN(eps=1e-6) -> fp32 residual stream X
// ---------------------------------------------------------------------------
__global__ __launch_bounds__(256) void embed_ln(
    const int* __restrict__ pos_sam, const int* __restrict__ neg_sam,
    const float* __restrict__ we, const float* __restrict__ pe,
    const float* __restrict__ w, const float* __restrict__ bsh,
    float* __restrict__ X)
{
    const int lane = threadIdx.x & 31;
    const int row  = blockIdx.x * 8 + (threadIdx.x >> 5);  // b*512+n
    const int n    = row & (GRID_N - 1);
    const int ps   = pos_sam[row];
    const int ns   = neg_sam[row];
    const int id   = (ns == 1) ? FALSE_LBL : ((ps == 1) ? TRUE_LBL : MASK_LBL);
    const float* wr = we + (size_t)id * DMODEL;
    const float* pr = pe + (size_t)n * DMODEL;
    float v[32], s = 0.0f, ss = 0.0f;
#pragma unroll
    for (int i = 0; i < 32; ++i) {
        const int d = i * 32 + lane;
        const float x = wr[d] + pr[d];
        v[i] = x; s += x; ss += x * x;
    }
    s  = rsum32(s);
    ss = rsum32(ss);
    const float mu  = s * (1.0f / DMODEL);
    const float var = ss * (1.0f / DMODEL) - mu * mu;
    const float rs  = rsqrtf(var + 1e-6f);
#pragma unroll
    for (int i = 0; i < 32; ++i) {
        const int d = i * 32 + lane;
        X[(size_t)row * DMODEL + d] = (v[i] - mu) * rs * w[d] + bsh[d];
    }
}

// ---------------------------------------------------------------------------
// Final LN + 1-logit head + sigmoid + BCE/metrics reduction (atomics)
// scratch: [loss_raw, pos_hit, pos_sum, neg_hit, neg_sum]
// ---------------------------------------------------------------------------
__global__ __launch_bounds__(256) void head_loss(
    const float* __restrict__ X,
    const float* __restrict__ lnw, const float* __restrict__ lnb,
    const float* __restrict__ ow,  const float* __restrict__ ob,
    const float* __restrict__ pos, const float* __restrict__ neg,
    float* __restrict__ scratch)
{
    const int lane = threadIdx.x & 31;
    const int row  = blockIdx.x * 8 + (threadIdx.x >> 5);
    const float* xr = X + (size_t)row * DMODEL;
    float v[32], s = 0.0f, ss = 0.0f;
#pragma unroll
    for (int i = 0; i < 32; ++i) {
        const float x = xr[i * 32 + lane];
        v[i] = x; s += x; ss += x * x;
    }
    s  = rsum32(s);
    ss = rsum32(ss);
    const float mu  = s * (1.0f / DMODEL);
    const float var = ss * (1.0f / DMODEL) - mu * mu;
    const float rs  = rsqrtf(var + 1e-5f);
    float dot = 0.0f;
#pragma unroll
    for (int i = 0; i < 32; ++i) {
        const int d = i * 32 + lane;
        dot += ((v[i] - mu) * rs * lnw[d] + lnb[d]) * ow[d];
    }
    dot = rsum32(dot);
    if (lane == 0) {
        const float z    = dot + ob[0];
        const float prob = 1.0f / (1.0f + __expf(-z));
        const float p    = fminf(fmaxf(prob, 1e-7f), 1.0f - 1e-7f);
        const float po   = pos[row];
        const float ne   = neg[row];
        const float lt   = -(po * __logf(p) + ne * __logf(1.0f - p));
        atomicAdd(&scratch[0], lt);
        atomicAdd(&scratch[1], (prob > 0.5f) ? po : 0.0f);
        atomicAdd(&scratch[2], po);
        atomicAdd(&scratch[3], ((1.0f - prob) > 0.5f) ? ne : 0.0f);
        atomicAdd(&scratch[4], ne);
    }
}

__global__ void zero_scratch(float* p) {
    if (threadIdx.x < 8) p[threadIdx.x] = 0.0f;
}

__global__ void finalize_out(const float* __restrict__ scratch,
                             float* __restrict__ out)
{
    if (threadIdx.x == 0) {
        const float denom = scratch[2] + scratch[4];
        out[0] = scratch[0] / denom;
        out[1] = scratch[1];
        out[2] = scratch[2];
        out[3] = scratch[3];
        out[4] = scratch[4];
    }
}

// ---------------------------------------------------------------------------
// Host orchestration
// ---------------------------------------------------------------------------
extern "C" void kernel_launch(void* const* d_in, const int* in_sizes, int n_in,
                              void* d_out, int out_size, void* d_ws, size_t ws_size,
                              hipStream_t stream)
{
    (void)in_sizes; (void)n_in; (void)out_size; (void)ws_size;

    const int*   pos_sam = (const int*)  d_in[0];
    const int*   neg_sam = (const int*)  d_in[1];
    const float* postive = (const float*)d_in[2];
    const float* negative= (const float*)d_in[3];
    const float* word_emb= (const float*)d_in[4];
    const float* pos_emb = (const float*)d_in[5];
    const float* lne_w   = (const float*)d_in[6];
    const float* lne_b   = (const float*)d_in[7];
    const float* ln1_w   = (const float*)d_in[8];
    const float* ln1_b   = (const float*)d_in[9];
    const float* qkv_w   = (const float*)d_in[10];
    const float* qkv_b   = (const float*)d_in[11];
    const float* proj_w  = (const float*)d_in[12];
    const float* proj_b  = (const float*)d_in[13];
    const float* ln2_w   = (const float*)d_in[14];
    const float* ln2_b   = (const float*)d_in[15];
    const float* fc1_w   = (const float*)d_in[16];
    const float* fc1_b   = (const float*)d_in[17];
    const float* fc2_w   = (const float*)d_in[18];
    const float* fc2_b   = (const float*)d_in[19];
    const float* lnf_w   = (const float*)d_in[20];
    const float* lnf_b   = (const float*)d_in[21];
    const float* out_w   = (const float*)d_in[22];
    const float* out_b   = (const float*)d_in[23];

    char*  ws  = (char*)d_ws;
    size_t off = 0;
    auto alloc = [&](size_t bytes) {
        void* p = ws + off;
        off += (bytes + 255) & ~(size_t)255;
        return p;
    };

    const size_t nQKVw = (size_t)NLAYER * 3 * DMODEL * DMODEL;   // 25.2M
    const size_t nPRJw = (size_t)NLAYER * DMODEL * DMODEL;       //  8.4M
    const size_t nFC1w = (size_t)NLAYER * FFDIM * DMODEL;        // 33.6M
    const size_t nFC2w = (size_t)NLAYER * DMODEL * FFDIM;        // 33.6M

    unsigned short* QKVw = (unsigned short*)alloc(nQKVw * 2);
    unsigned short* PRJw = (unsigned short*)alloc(nPRJw * 2);
    unsigned short* FC1w = (unsigned short*)alloc(nFC1w * 2);
    unsigned short* FC2w = (unsigned short*)alloc(nFC2w * 2);
    float*          X     = (float*)         alloc((size_t)MROWS * DMODEL * 4);
    unsigned short* Hbf   = (unsigned short*)alloc((size_t)MROWS * DMODEL * 2);
    float*          QKVf  = (float*)         alloc((size_t)MROWS * 3 * DMODEL * 4);
    unsigned short* Qb    = (unsigned short*)alloc((size_t)BATCH * NHEAD * GRID_N * HEADD * 2);
    unsigned short* Kb    = (unsigned short*)alloc((size_t)BATCH * NHEAD * GRID_N * HEADD * 2);
    unsigned short* Vt    = (unsigned short*)alloc((size_t)BATCH * NHEAD * GRID_N * HEADD * 2);
    unsigned short* Ybf   = (unsigned short*)alloc((size_t)MROWS * DMODEL * 2);
    unsigned short* H2bf  = (unsigned short*)alloc((size_t)MROWS * DMODEL * 2);
    unsigned short* Gbf   = (unsigned short*)alloc((size_t)MROWS * FFDIM * 2);
    float*          scr   = (float*)         alloc(64 * 4);

    zero_scratch<<<1, 32, 0, stream>>>(scr);

    // one-time fp32 -> bf16 weight conversion (L2-resident afterwards)
    conv_f32_bf16<<<(int)((nQKVw / 4 + 255) / 256), 256, 0, stream>>>(qkv_w, QKVw, (int)(nQKVw / 4));
    conv_f32_bf16<<<(int)((nPRJw / 4 + 255) / 256), 256, 0, stream>>>(proj_w, PRJw, (int)(nPRJw / 4));
    conv_f32_bf16<<<(int)((nFC1w / 4 + 255) / 256), 256, 0, stream>>>(fc1_w, FC1w, (int)(nFC1w / 4));
    conv_f32_bf16<<<(int)((nFC2w / 4 + 255) / 256), 256, 0, stream>>>(fc2_w, FC2w, (int)(nFC2w / 4));

    embed_ln<<<MROWS / 8, 256, 0, stream>>>(pos_sam, neg_sam, word_emb, pos_emb,
                                            lne_w, lne_b, X);

    for (int l = 0; l < NLAYER; ++l) {
        ln_to_bf16<<<MROWS / 8, 256, 0, stream>>>(
            X, ln1_w + (size_t)l * DMODEL, ln1_b + (size_t)l * DMODEL, 1e-5f, Hbf);
        gemm_bf16_wmma<0><<<dim3(3 * DMODEL / BN, MROWS / BM), 256, 0, stream>>>(
            Hbf, QKVw + (size_t)l * 3 * DMODEL * DMODEL,
            qkv_b + (size_t)l * 3 * DMODEL, nullptr, QKVf, nullptr,
            MROWS, 3 * DMODEL, DMODEL);
        qkv_reorg<<<(BATCH * NHEAD * GRID_N * HEADD) / 256, 256, 0, stream>>>(
            QKVf, Qb, Kb, Vt);
        attn_flash<<<(BATCH * NHEAD * (GRID_N / 16)) / 8, 256, 0, stream>>>(
            Qb, Kb, Vt, Ybf);
        gemm_bf16_wmma<1><<<dim3(DMODEL / BN, MROWS / BM), 256, 0, stream>>>(
            Ybf, PRJw + (size_t)l * DMODEL * DMODEL,
            proj_b + (size_t)l * DMODEL, X, X, nullptr,
            MROWS, DMODEL, DMODEL);
        ln_to_bf16<<<MROWS / 8, 256, 0, stream>>>(
            X, ln2_w + (size_t)l * DMODEL, ln2_b + (size_t)l * DMODEL, 1e-5f, H2bf);
        gemm_bf16_wmma<2><<<dim3(FFDIM / BN, MROWS / BM), 256, 0, stream>>>(
            H2bf, FC1w + (size_t)l * FFDIM * DMODEL,
            fc1_b + (size_t)l * FFDIM, nullptr, nullptr, Gbf,
            MROWS, FFDIM, DMODEL);
        gemm_bf16_wmma<1><<<dim3(DMODEL / BN, MROWS / BM), 256, 0, stream>>>(
            Gbf, FC2w + (size_t)l * DMODEL * FFDIM,
            fc2_b + (size_t)l * DMODEL, X, X, nullptr,
            MROWS, DMODEL, FFDIM);
    }

    head_loss<<<MROWS / 8, 256, 0, stream>>>(X, lnf_w, lnf_b, out_w, out_b,
                                             postive, negative, scr);
    finalize_out<<<1, 32, 0, stream>>>(scr, (float*)d_out);
}
